// GemmaAttention_24799141167410
// MI455X (gfx1250) — compile-verified, matrix operand order checked by reading, not run on hardware
//
#include <hip/hip_runtime.h>

// ---------------- problem constants ----------------
#define SLEN 2048
#define HID  3584
#define NH   16
#define NKV  8
#define HD   256
#define QKVO ((NH + 2*NKV) * HD)   // 8192
#define WINDOW 1024
#define SOFTCAP 50.0f
#define SCALING 0.0625f            // 256^-0.5
#define EPS 1e-6f

typedef __attribute__((ext_vector_type(16))) __bf16 v16bf;
typedef __attribute__((ext_vector_type(8)))  __bf16 v8bf;
typedef __attribute__((ext_vector_type(8)))  float  v8f;

// exact parameter types for the gfx1250 async-to-LDS builtin (per clang diagnostic)
typedef int i32x4 __attribute__((vector_size(16)));
typedef __attribute__((address_space(1))) i32x4 gbl_i32x4;   // global ("__device__")
typedef __attribute__((address_space(3))) i32x4 lds_i32x4;   // LDS ("__shared__")

static __device__ __forceinline__ v16bf make16(v8bf lo, v8bf hi) {
  return __builtin_shufflevector(lo, hi, 0,1,2,3,4,5,6,7,8,9,10,11,12,13,14,15);
}
static __device__ __forceinline__ v8f wmma_bf16(v16bf a, v16bf b, v8f c) {
  return __builtin_amdgcn_wmma_f32_16x16x32_bf16(false, a, false, b, (short)0, c, false, false);
}

// --------- gfx1250 async global->LDS copy (ASYNCcnt path), guarded ---------
static __device__ __forceinline__ void async_copy16(const __bf16* g, __bf16* l) {
#if __has_builtin(__builtin_amdgcn_global_load_async_to_lds_b128)
  __builtin_amdgcn_global_load_async_to_lds_b128(
      (gbl_i32x4*)g, (lds_i32x4*)l, 0, 0);
#else
  *(v8bf*)l = *(const v8bf*)g;   // fallback: sync copy through VGPRs
#endif
}
static __device__ __forceinline__ void async_wait0() {
#if __has_builtin(__builtin_amdgcn_s_wait_asynccnt)
  __builtin_amdgcn_s_wait_asynccnt(0);
#elif __has_builtin(__builtin_amdgcn_global_load_async_to_lds_b128)
  asm volatile("s_wait_asynccnt 0" ::: "memory");
#endif
}

// stable fast softcap: SOFTCAP * tanh(s*SCALING/SOFTCAP)
static __device__ __forceinline__ float softcap_score(float s) {
  float x = s * (SCALING / SOFTCAP);
  float t = __expf(-2.0f * fabsf(x));
  float r = (1.0f - t) / (1.0f + t);
  return SOFTCAP * copysignf(r, x);
}

// ---------------- 1) fp32 -> bf16 convert ----------------
__global__ __launch_bounds__(256)
void cvt_f32_bf16(const float* __restrict__ src, __bf16* __restrict__ dst, size_t n) {
  size_t i = ((size_t)blockIdx.x * blockDim.x + threadIdx.x) * 8;
  if (i + 8 <= n) {
    float4 a = *(const float4*)(src + i);
    float4 b = *(const float4*)(src + i + 4);
    v8bf v = { (__bf16)a.x, (__bf16)a.y, (__bf16)a.z, (__bf16)a.w,
               (__bf16)b.x, (__bf16)b.y, (__bf16)b.z, (__bf16)b.w };
    *(v8bf*)(dst + i) = v;
  }
}

// ---------------- 2) fp32 [R][C] -> bf16 [C][R] transpose+convert ----------------
__global__ __launch_bounds__(256)
void transpose_cvt(const float* __restrict__ src, __bf16* __restrict__ dst, int R, int C) {
  __shared__ float t[32][33];
  int c0 = blockIdx.x * 32, r0 = blockIdx.y * 32;
  int x = threadIdx.x, y = threadIdx.y;         // 32 x 8
#pragma unroll
  for (int j = 0; j < 4; ++j)
    t[y + j*8][x] = src[(size_t)(r0 + y + j*8) * C + c0 + x];
  __syncthreads();
#pragma unroll
  for (int j = 0; j < 4; ++j)
    dst[(size_t)(c0 + y + j*8) * R + r0 + x] = (__bf16)t[x][y + j*8];
}

// ---------------- 3) bf16 GEMM: C[M][N] = A[M][K] * BT[N][K]^T ----------------
// 128x128x32 tiles, 8 waves (4 along M x 2 along N), double-buffered async LDS.
// Pipeline: wait-at-top for loads issued one iteration ago; issue next tile
// right after the barrier so DMA overlaps the whole WMMA block.
#define GSTR 40
__global__ __launch_bounds__(256)
void gemm_bf16_tn(const __bf16* __restrict__ A, const __bf16* __restrict__ BT,
                  float* __restrict__ C, int M, int N, int K) {
  __shared__ __bf16 As[2][128 * GSTR];
  __shared__ __bf16 Bs[2][128 * GSTR];
  const int mBlk = blockIdx.y * 128, nBlk = blockIdx.x * 128;
  const int tid = threadIdx.x;
  const int wave = tid >> 5, lane = tid & 31;
  const int half = lane >> 4, l16 = lane & 15;
  const int waveM = wave & 3;
  const int waveN = wave >> 2;

  auto loadTiles = [&](int k0, int buf) {
#pragma unroll
    for (int c = tid; c < 512; c += 256) {
      int row = c >> 2, col = (c & 3) * 8;
      async_copy16(A + (size_t)(mBlk + row) * K + k0 + col, &As[buf][row * GSTR + col]);
      async_copy16(BT + (size_t)(nBlk + row) * K + k0 + col, &Bs[buf][row * GSTR + col]);
      if (k0 + 64 < K) {  // L2 prefetch two tiles ahead
        __builtin_prefetch(A + (size_t)(mBlk + row) * K + k0 + 64 + col, 0, 2);
        __builtin_prefetch(BT + (size_t)(nBlk + row) * K + k0 + 64 + col, 0, 2);
      }
    }
  };

  v8f acc[2][4];
#pragma unroll
  for (int i = 0; i < 2; ++i)
#pragma unroll
    for (int j = 0; j < 4; ++j)
      acc[i][j] = (v8f){0,0,0,0,0,0,0,0};

  loadTiles(0, 0);

  for (int k0 = 0; k0 < K; k0 += 32) {
    const int cur = (k0 >> 5) & 1;

    // loads issued last iteration (for `cur`) have overlapped its compute
    async_wait0();
    __syncthreads();

    if (k0 + 32 < K) loadTiles(k0 + 32, cur ^ 1);   // overlaps WMMAs below

    v16bf aF[2], bF[4];
#pragma unroll
    for (int i = 0; i < 2; ++i) {
      int r = waveM * 32 + i * 16 + l16;
      v8bf lo = *(const v8bf*)(&As[cur][r * GSTR + half * 8]);
      v8bf hi = *(const v8bf*)(&As[cur][r * GSTR + 16 + half * 8]);
      aF[i] = make16(lo, hi);
    }
#pragma unroll
    for (int j = 0; j < 4; ++j) {
      int r = waveN * 64 + j * 16 + l16;
      v8bf lo = *(const v8bf*)(&Bs[cur][r * GSTR + half * 8]);
      v8bf hi = *(const v8bf*)(&Bs[cur][r * GSTR + 16 + half * 8]);
      bF[j] = make16(lo, hi);
    }
#pragma unroll
    for (int i = 0; i < 2; ++i)
#pragma unroll
      for (int j = 0; j < 4; ++j)
        acc[i][j] = wmma_bf16(aF[i], bF[j], acc[i][j]);
  }

#pragma unroll
  for (int i = 0; i < 2; ++i)
#pragma unroll
    for (int j = 0; j < 4; ++j)
#pragma unroll
      for (int r = 0; r < 8; ++r) {
        int row = mBlk + waveM * 32 + i * 16 + half * 8 + r;
        int col = nBlk + waveN * 64 + j * 16 + l16;
        C[(size_t)row * N + col] = acc[i][j][r];
      }
}

// ---------------- 4) fused RMS-norm + RoPE for Q and K ----------------
__global__ __launch_bounds__(256)
void qk_norm_rope(const float* __restrict__ qkvf,
                  const float* __restrict__ qnw, const float* __restrict__ knw,
                  const float* __restrict__ cosb, const float* __restrict__ sinb,
                  __bf16* __restrict__ qb, __bf16* __restrict__ kb) {
  const int wave = threadIdx.x >> 5, lane = threadIdx.x & 31;
  const int s = blockIdx.x * 8 + wave;
  const int head = blockIdx.y;
  const bool isQ = head < NH;
  const float* w = isQ ? qnw : knw;
  const float* src = qkvf + (size_t)s * QKVO + head * HD;

  float xr[8], ss = 0.f;
#pragma unroll
  for (int i = 0; i < 8; ++i) { xr[i] = src[lane + 32 * i]; ss += xr[i] * xr[i]; }
#pragma unroll
  for (int off = 16; off > 0; off >>= 1) ss += __shfl_xor(ss, off, 32);
  const float inv = rsqrtf(ss * (1.0f / HD) + EPS);

  float xn[8];
#pragma unroll
  for (int i = 0; i < 8; ++i) {
    int d = lane + 32 * i;
    xn[i] = xr[i] * inv * (1.0f + w[d]);
  }
  __bf16* dst = isQ ? (qb + ((size_t)head * SLEN + s) * HD)
                    : (kb + ((size_t)(head - NH) * SLEN + s) * HD);
#pragma unroll
  for (int i = 0; i < 8; ++i) {
    int d = lane + 32 * i;
    float rot = (i < 4) ? -xn[i + 4] : xn[i - 4];
    float o = xn[i] * cosb[(size_t)s * HD + d] + rot * sinb[(size_t)s * HD + d];
    dst[d] = (__bf16)o;
  }
}

// ---------------- 5) V: slice, convert, transpose to [kv][d][s] bf16 ----------------
__global__ __launch_bounds__(256)
void v_transpose(const float* __restrict__ qkvf, __bf16* __restrict__ vtb) {
  __shared__ float t[32][33];
  int s0 = blockIdx.x * 32, d0 = blockIdx.y * 32, kv = blockIdx.z;
  int x = threadIdx.x, y = threadIdx.y;
  int colBase = (NH + NKV) * HD + kv * HD + d0;
#pragma unroll
  for (int j = 0; j < 4; ++j)
    t[y + j*8][x] = qkvf[(size_t)(s0 + y + j*8) * QKVO + colBase + x];
  __syncthreads();
#pragma unroll
  for (int j = 0; j < 4; ++j)
    vtb[((size_t)kv * HD + d0 + y + j*8) * SLEN + s0 + x] = (__bf16)t[x][y + j*8];
}

// ---------------- 6) flash attention, sliding window + softcap ----------------
__global__ __launch_bounds__(256)
void attention(const __bf16* __restrict__ qb, const __bf16* __restrict__ kb,
               const __bf16* __restrict__ vtb, __bf16* __restrict__ aob) {
  __shared__ __bf16 Ks[2][32][264];
  __shared__ __bf16 Vs[2][HD][40];
  __shared__ __bf16 Ps[8][16][40];

  const int h = blockIdx.y, kv = h / (NH / NKV);
  const int q0 = blockIdx.x * 128;
  const int tid = threadIdx.x, wave = tid >> 5, lane = tid & 31;
  const int half = lane >> 4, l16 = lane & 15;
  const int qRow = q0 + wave * 16;

  auto loadKV = [&](int j0, int buf) {
#pragma unroll
    for (int c = tid; c < 1024; c += 256) {
      int rk = c >> 5, ck = (c & 31) * 8;
      async_copy16(kb + ((size_t)kv * SLEN + j0 + rk) * HD + ck, &Ks[buf][rk][ck]);
      int rv = c >> 2, cvx = (c & 3) * 8;
      async_copy16(vtb + ((size_t)kv * HD + rv) * SLEN + j0 + cvx, &Vs[buf][rv][cvx]);
    }
  };

  // resident Q fragments (A layout)
  v16bf qF[8];
  {
    const __bf16* qp = qb + ((size_t)h * SLEN + qRow + l16) * HD;
#pragma unroll
    for (int c = 0; c < 8; ++c) {
      v8bf lo = *(const v8bf*)(qp + c * 32 + half * 8);
      v8bf hi = *(const v8bf*)(qp + c * 32 + 16 + half * 8);
      qF[c] = make16(lo, hi);
    }
  }

  v8f o[16];
#pragma unroll
  for (int nt = 0; nt < 16; ++nt) o[nt] = (v8f){0,0,0,0,0,0,0,0};
  float mrow[8], lrow[8];
#pragma unroll
  for (int r = 0; r < 8; ++r) { mrow[r] = -1e30f; lrow[r] = 0.f; }

  int jlo = q0 - (WINDOW - 1); if (jlo < 0) jlo = 0;
  const int t0 = (jlo / 32) * 32;
  const int t1 = q0 + 128;

  loadKV(t0, 0);

  for (int j0 = t0; j0 < t1; j0 += 32) {
    const int cur = ((j0 - t0) >> 5) & 1;

    async_wait0();       // loads for `cur` (issued last iter) are done
    __syncthreads();

    if (j0 + 32 < t1) loadKV(j0 + 32, cur ^ 1);   // overlap with compute

    v8f sAcc[2];
    sAcc[0] = (v8f){0,0,0,0,0,0,0,0};
    sAcc[1] = (v8f){0,0,0,0,0,0,0,0};
#pragma unroll
    for (int c = 0; c < 8; ++c) {
#pragma unroll
      for (int ni = 0; ni < 2; ++ni) {
        int r = ni * 16 + l16;
        v8bf lo = *(const v8bf*)(&Ks[cur][r][c * 32 + half * 8]);
        v8bf hi = *(const v8bf*)(&Ks[cur][r][c * 32 + 16 + half * 8]);
        sAcc[ni] = wmma_bf16(qF[c], make16(lo, hi), sAcc[ni]);
      }
    }

    float p[2][8];
#pragma unroll
    for (int ni = 0; ni < 2; ++ni)
#pragma unroll
      for (int r = 0; r < 8; ++r) {
        float sc = softcap_score(sAcc[ni][r]);
        int j = j0 + ni * 16 + l16;
        int i = qRow + half * 8 + r;
        bool ok = (j <= i) && (i - j < WINDOW);
        p[ni][r] = ok ? sc : -__builtin_inff();
      }

    float scale[8];
#pragma unroll
    for (int r = 0; r < 8; ++r) {
      float t = fmaxf(p[0][r], p[1][r]);
#pragma unroll
      for (int off = 8; off > 0; off >>= 1) t = fmaxf(t, __shfl_xor(t, off, 32));
      float nm = fmaxf(mrow[r], t);
      scale[r] = __expf(mrow[r] - nm);
      mrow[r] = nm;
      float rs = 0.f;
#pragma unroll
      for (int ni = 0; ni < 2; ++ni) { p[ni][r] = __expf(p[ni][r] - nm); rs += p[ni][r]; }
#pragma unroll
      for (int off = 8; off > 0; off >>= 1) rs += __shfl_xor(rs, off, 32);
      lrow[r] = lrow[r] * scale[r] + rs;
    }
#pragma unroll
    for (int nt = 0; nt < 16; ++nt)
#pragma unroll
      for (int r = 0; r < 8; ++r) o[nt][r] *= scale[r];

    // re-layout P: C layout -> A layout via per-wave LDS
#pragma unroll
    for (int ni = 0; ni < 2; ++ni)
#pragma unroll
      for (int r = 0; r < 8; ++r)
        Ps[wave][half * 8 + r][ni * 16 + l16] = (__bf16)p[ni][r];
    v16bf pF;
    {
      v8bf lo = *(const v8bf*)(&Ps[wave][l16][half * 8]);
      v8bf hi = *(const v8bf*)(&Ps[wave][l16][16 + half * 8]);
      pF = make16(lo, hi);
    }

#pragma unroll
    for (int nt = 0; nt < 16; ++nt) {
      v8bf lo = *(const v8bf*)(&Vs[cur][nt * 16 + l16][half * 8]);
      v8bf hi = *(const v8bf*)(&Vs[cur][nt * 16 + l16][16 + half * 8]);
      o[nt] = wmma_bf16(pF, make16(lo, hi), o[nt]);
    }
  }

#pragma unroll
  for (int nt = 0; nt < 16; ++nt)
#pragma unroll
    for (int r = 0; r < 8; ++r) {
      int srow = qRow + half * 8 + r;
      int dcol = nt * 16 + l16;
      aob[(size_t)srow * (NH * HD) + h * HD + dcol] = (__bf16)(o[nt][r] / lrow[r]);
    }
}

// ---------------- host: orchestrate ----------------
extern "C" void kernel_launch(void* const* d_in, const int* in_sizes, int n_in,
                              void* d_out, int out_size, void* d_ws, size_t ws_size,
                              hipStream_t stream) {
  (void)in_sizes; (void)n_in; (void)out_size; (void)ws_size;
  const float* hs    = (const float*)d_in[0];
  const float* w_qkv = (const float*)d_in[1];
  const float* w_o   = (const float*)d_in[2];
  const float* qnw   = (const float*)d_in[3];
  const float* knw   = (const float*)d_in[4];
  const float* cosb  = (const float*)d_in[5];
  const float* sinb  = (const float*)d_in[6];
  float* out = (float*)d_out;

  char* ws = (char*)d_ws;
  size_t off = 0;
  auto take = [&](size_t bytes) { char* p = ws + off; off += (bytes + 255) & ~(size_t)255; return p; };
  __bf16* hsb   = (__bf16*)take((size_t)SLEN * HID * 2);
  __bf16* wqkvT = (__bf16*)take((size_t)QKVO * HID * 2);
  float*  qkvf  = (float*) take((size_t)SLEN * QKVO * 4);
  __bf16* qb    = (__bf16*)take((size_t)NH  * SLEN * HD * 2);
  __bf16* kb    = (__bf16*)take((size_t)NKV * SLEN * HD * 2);
  __bf16* vtb   = (__bf16*)take((size_t)NKV * HD * SLEN * 2);
  __bf16* aob   = (__bf16*)take((size_t)SLEN * NH * HD * 2);
  __bf16* woT   = (__bf16*)take((size_t)HID * NH * HD * 2);

  {
    size_t n = (size_t)SLEN * HID;
    cvt_f32_bf16<<<(unsigned)(n / 8 / 256), 256, 0, stream>>>(hs, hsb, n);
  }
  transpose_cvt<<<dim3(QKVO / 32, HID / 32), dim3(32, 8), 0, stream>>>(w_qkv, wqkvT, HID, QKVO);
  transpose_cvt<<<dim3(HID / 32, (NH * HD) / 32), dim3(32, 8), 0, stream>>>(w_o, woT, NH * HD, HID);
  gemm_bf16_tn<<<dim3(QKVO / 128, SLEN / 128), 256, 0, stream>>>(hsb, wqkvT, qkvf, SLEN, QKVO, HID);
  qk_norm_rope<<<dim3(SLEN / 8, NH + NKV), 256, 0, stream>>>(qkvf, qnw, knw, cosb, sinb, qb, kb);
  v_transpose<<<dim3(SLEN / 32, HD / 32, NKV), dim3(32, 8), 0, stream>>>(qkvf, vtb);
  attention<<<dim3(SLEN / 128, NH), 256, 0, stream>>>(qb, kb, vtb, aob);
  gemm_bf16_tn<<<dim3(HID / 128, SLEN / 128), 256, 0, stream>>>(aob, woT, out, SLEN, HID, NH * HD);
}
